// FeaturePropagation_65120294142110
// MI455X (gfx1250) — matrix operand order repaired and checked by reading, not compiled, not run on hardware
//
#include <hip/hip_runtime.h>

// Problem sizes (fixed by the reference)
#define B_   4
#define N_   16384
#define M_   4096
#define D1_  128
#define D2_  256
#define CIN_ 384          // D1 + D2
#define C1_  256
#define C2_  128
#define NP_  (B_ * N_)    // 65536 points total

typedef _Float16 v16h __attribute__((ext_vector_type(16)));
typedef _Float16 v8h  __attribute__((ext_vector_type(8)));
typedef float    v8f  __attribute__((ext_vector_type(8)));

__device__ __forceinline__ v16h make_v16(v8h lo, v8h hi) {
  return __builtin_shufflevector(lo, hi, 0, 1, 2, 3, 4, 5, 6, 7,
                                          8, 9, 10, 11, 12, 13, 14, 15);
}

// ---------------------------------------------------------------------------
// Stage 1: branchless 3-NN over LDS-resident xyz2 (48 KB per batch)
// ---------------------------------------------------------------------------
__global__ void __launch_bounds__(256)
knn3_kernel(const float* __restrict__ xyz1, const float* __restrict__ xyz2,
            int* __restrict__ idx_out, float* __restrict__ w_out) {
  __shared__ float s[M_ * 3];
  const int b = blockIdx.y;
  const float* src = xyz2 + (size_t)b * M_ * 3;
  for (int i = threadIdx.x; i < M_ * 3; i += 256) s[i] = src[i];
  __syncthreads();

  const int q = blockIdx.x * 256 + threadIdx.x;
  const float* p = xyz1 + ((size_t)b * N_ + q) * 3;
  const float qx = p[0], qy = p[1], qz = p[2];

  float d0 = 3.4e38f, d1 = 3.4e38f, d2 = 3.4e38f;
  int   i0 = 0, i1 = 0, i2 = 0;
  for (int j = 0; j < M_; ++j) {
    const float dx = qx - s[3 * j + 0];
    const float dy = qy - s[3 * j + 1];
    const float dz = qz - s[3 * j + 2];
    const float d  = fmaf(dx, dx, fmaf(dy, dy, dz * dz));
    const bool c0 = d < d0, c1 = d < d1, c2 = d < d2;
    const float nd2 = c1 ? d1 : (c2 ? d : d2); const int ni2 = c1 ? i1 : (c2 ? j : i2);
    const float nd1 = c0 ? d0 : (c1 ? d : d1); const int ni1 = c0 ? i0 : (c1 ? j : i1);
    const float nd0 = c0 ? d  : d0;            const int ni0 = c0 ? j  : i0;
    d0 = nd0; d1 = nd1; d2 = nd2; i0 = ni0; i1 = ni1; i2 = ni2;
  }
  const float w0 = 1.f / fmaxf(d0, 1e-10f);
  const float w1 = 1.f / fmaxf(d1, 1e-10f);
  const float w2 = 1.f / fmaxf(d2, 1e-10f);
  const float inv = 1.f / (w0 + w1 + w2);
  const size_t o = ((size_t)b * N_ + q) * 3;
  idx_out[o + 0] = i0; idx_out[o + 1] = i1; idx_out[o + 2] = i2;
  w_out[o + 0] = w0 * inv; w_out[o + 1] = w1 * inv; w_out[o + 2] = w2 * inv;
}

// ---------------------------------------------------------------------------
// Stage 2: gather + interpolate + concat -> X1 [NP, 384] as f16 (GEMM A)
// One block per point: 128 threads; t -> points1 channel t, interp ch 2t,2t+1
// ---------------------------------------------------------------------------
__global__ void __launch_bounds__(128)
interp_concat_kernel(const float* __restrict__ points1,
                     const float* __restrict__ points2,
                     const int* __restrict__ idx, const float* __restrict__ wgt,
                     _Float16* __restrict__ X1) {
  const int p = blockIdx.x;       // 0..NP-1 == b*N + n
  const int b = p >> 14;          // N_ == 2^14
  const int t = threadIdx.x;
  _Float16* out = X1 + (size_t)p * CIN_;

  out[t] = (_Float16)points1[(size_t)p * D1_ + t];

  const int*   id = idx + (size_t)p * 3;
  const float* w  = wgt + (size_t)p * 3;
  const float* base = points2 + (size_t)b * M_ * D2_;
  float ax = 0.f, ay = 0.f;
#pragma unroll
  for (int k = 0; k < 3; ++k) {
    const float2 v = *(const float2*)(base + (size_t)id[k] * D2_ + 2 * t);
    ax = fmaf(w[k], v.x, ax);
    ay = fmaf(w[k], v.y, ay);
  }
  out[D1_ + 2 * t + 0] = (_Float16)ax;
  out[D1_ + 2 * t + 1] = (_Float16)ay;
}

// ---------------------------------------------------------------------------
// f32 -> f16 weight conversion
// ---------------------------------------------------------------------------
__global__ void cvt_f32_f16(const float* __restrict__ in, _Float16* __restrict__ out, int n) {
  const int i = blockIdx.x * blockDim.x + threadIdx.x;
  if (i < n) out[i] = (_Float16)in[i];
}

// ---------------------------------------------------------------------------
// WMMA GEMM: Hout[NP, NOUT] = A[NP, K](f16) * Wt[NOUT, K](f16)^T + bias (f32)
// Block: 128 threads = 4 waves; 64x64 tile; each wave -> 2x2 WMMA 16x16 tiles.
// Tile staging: two global_load_b128 + two ds_store_b128 per thread per tile
// (16B-aligned in global: row pitch 768/512 B; in LDS: 80 B padded pitch).
// LDS row pitch = 40 f16 (80 B) -> 16B-aligned, bank-spread b128 frag loads.
// ---------------------------------------------------------------------------
template <int K, int NOUT>
__global__ void __launch_bounds__(128)
gemm_f16_wmma(const _Float16* __restrict__ A, const _Float16* __restrict__ Wt,
              const float* __restrict__ bias, float* __restrict__ Hout) {
  __shared__ _Float16 sA[64 * 40];
  __shared__ _Float16 sB[64 * 40];

  const int m0   = blockIdx.x * 64;
  const int n0   = blockIdx.y * 64;
  const int tid  = threadIdx.x;
  const int lane = tid & 31;
  const int wv   = tid >> 5;
  const int wm   = (wv >> 1) * 32;  // wave M offset in tile
  const int wn   = (wv & 1) * 32;   // wave N offset in tile
  const int lr   = lane & 15;
  const int hs   = lane >> 4;       // half-wave select

  v8f acc[2][2] = {};

  for (int k0 = 0; k0 < K; k0 += 32) {
    // Cooperative tile staging: 64 rows x 32 f16 = 4 KB per matrix per step.
    // 256 uint4 chunks / 128 threads -> 2 chunks each (A and B).
#pragma unroll
    for (int it = 0; it < 2; ++it) {
      const int e = tid + it * 128;        // 0..255
      const int row = e >> 2, qd = e & 3;  // quarter-row of 8 f16 (16 B)
      *(uint4*)&sA[row * 40 + qd * 8] =
          *(const uint4*)(A + (size_t)(m0 + row) * K + k0 + qd * 8);
      *(uint4*)&sB[row * 40 + qd * 8] =
          *(const uint4*)(Wt + (size_t)(n0 + row) * K + k0 + qd * 8);
    }
    __syncthreads();

    // A-fragments (ISA 16-bit A 16x32): lane<16 -> K{0..7,16..23}, lane>=16 -> +8
    v16h afr[2], bfr[2];
#pragma unroll
    for (int mt = 0; mt < 2; ++mt) {
      const _Float16* pa = &sA[(wm + mt * 16 + lr) * 40 + hs * 8];
      afr[mt] = make_v16(*(const v8h*)pa, *(const v8h*)(pa + 16));
    }
    // B-fragments (dense 32x16): lane<16 -> K{0..7,16..23}, lane>=16 -> +8
#pragma unroll
    for (int nt = 0; nt < 2; ++nt) {
      const _Float16* pb = &sB[(wn + nt * 16 + lr) * 40 + hs * 8];
      bfr[nt] = make_v16(*(const v8h*)pb, *(const v8h*)(pb + 16));
    }

#pragma unroll
    for (int mt = 0; mt < 2; ++mt)
#pragma unroll
      for (int nt = 0; nt < 2; ++nt)
        acc[mt][nt] = __builtin_amdgcn_wmma_f32_16x16x32_f16(
            false, afr[mt], false, bfr[nt], (short)0, acc[mt][nt], false, false);

    __syncthreads();
  }

  // Epilogue: D layout — lane<16: N=lane, M=r; lane>=16: N=lane-16, M=r+8
#pragma unroll
  for (int mt = 0; mt < 2; ++mt) {
#pragma unroll
    for (int nt = 0; nt < 2; ++nt) {
      const int gn = n0 + wn + nt * 16 + lr;
      const int gm = m0 + wm + mt * 16 + hs * 8;
      const float bia = bias[gn];
#pragma unroll
      for (int r = 0; r < 8; ++r)
        Hout[(size_t)(gm + r) * NOUT + gn] = acc[mt][nt][r] + bia;
    }
  }
}

// ---------------------------------------------------------------------------
// Deterministic per-channel sum / sum-of-squares (two-stage, no float atomics)
// ---------------------------------------------------------------------------
template <int NOUT>
__global__ void colstat_partial(const float* __restrict__ H, float* __restrict__ part) {
  const int c = threadIdx.x;                          // blockDim.x == NOUT
  const size_t r0 = (size_t)blockIdx.x * (NP_ / 256); // 256 blocks x 256 rows
  float s = 0.f, s2 = 0.f;
  for (int r = 0; r < NP_ / 256; ++r) {
    const float v = H[(r0 + r) * NOUT + c];
    s += v;
    s2 = fmaf(v, v, s2);
  }
  part[(size_t)blockIdx.x * (2 * NOUT) + c] = s;
  part[(size_t)blockIdx.x * (2 * NOUT) + NOUT + c] = s2;
}

template <int NOUT>
__global__ void colstat_final(const float* __restrict__ part, float* __restrict__ stats) {
  __shared__ float red[256];
  const int t = threadIdx.x;    // 256
  const int j = blockIdx.x;     // 0..2*NOUT-1 (sum then sumsq per channel)
  red[t] = part[(size_t)t * (2 * NOUT) + j];
  __syncthreads();
  for (int sft = 128; sft > 0; sft >>= 1) {
    if (t < sft) red[t] += red[t + sft];
    __syncthreads();
  }
  if (t == 0) stats[j] = red[0];
}

// ---------------------------------------------------------------------------
// BN (train-mode, biased var) + ReLU epilogues
// ---------------------------------------------------------------------------
__global__ void bn_relu_to_f16(const float* __restrict__ H, const float* __restrict__ stats,
                               const float* __restrict__ g, const float* __restrict__ be,
                               _Float16* __restrict__ out, int nout, int total) {
  const int i = blockIdx.x * blockDim.x + threadIdx.x;
  if (i >= total) return;
  const int c = i & (nout - 1);
  const float inv  = 1.0f / (float)NP_;
  const float mean = stats[c] * inv;
  const float var  = stats[nout + c] * inv - mean * mean;
  const float y = g[c] * (H[i] - mean) * rsqrtf(var + 1e-5f) + be[c];
  out[i] = (_Float16)fmaxf(y, 0.f);
}

__global__ void bn_relu_inplace_f32(float* __restrict__ H, const float* __restrict__ stats,
                                    const float* __restrict__ g, const float* __restrict__ be,
                                    int nout, int total) {
  const int i = blockIdx.x * blockDim.x + threadIdx.x;
  if (i >= total) return;
  const int c = i & (nout - 1);
  const float inv  = 1.0f / (float)NP_;
  const float mean = stats[c] * inv;
  const float var  = stats[nout + c] * inv - mean * mean;
  const float y = g[c] * (H[i] - mean) * rsqrtf(var + 1e-5f) + be[c];
  H[i] = fmaxf(y, 0.f);
}

// ---------------------------------------------------------------------------
extern "C" void kernel_launch(void* const* d_in, const int* in_sizes, int n_in,
                              void* d_out, int out_size, void* d_ws, size_t ws_size,
                              hipStream_t stream) {
  const float* xyz1    = (const float*)d_in[0];
  const float* xyz2    = (const float*)d_in[1];
  const float* points1 = (const float*)d_in[2];
  const float* points2 = (const float*)d_in[3];
  const float* w1      = (const float*)d_in[4];
  const float* b1      = (const float*)d_in[5];
  const float* g1      = (const float*)d_in[6];
  const float* be1     = (const float*)d_in[7];
  const float* w2      = (const float*)d_in[8];
  const float* b2      = (const float*)d_in[9];
  const float* g2      = (const float*)d_in[10];
  const float* be2     = (const float*)d_in[11];
  float* out = (float*)d_out;

  char* ws = (char*)d_ws;
  size_t off = 0;
  auto take = [&](size_t bytes) -> void* {
    void* p = ws + off;
    off = (off + bytes + 255) & ~(size_t)255;
    return p;
  };
  int*      d_idx = (int*)      take((size_t)NP_ * 3 * sizeof(int));
  float*    d_wgt = (float*)    take((size_t)NP_ * 3 * sizeof(float));
  _Float16* d_X1  = (_Float16*) take((size_t)NP_ * CIN_ * sizeof(_Float16));
  _Float16* d_W1h = (_Float16*) take((size_t)C1_ * CIN_ * sizeof(_Float16));
  _Float16* d_W2h = (_Float16*) take((size_t)C2_ * C1_ * sizeof(_Float16));
  float*    d_H1  = (float*)    take((size_t)NP_ * C1_ * sizeof(float));
  _Float16* d_H1h = (_Float16*) take((size_t)NP_ * C1_ * sizeof(_Float16));
  float*    d_p1  = (float*)    take((size_t)256 * 2 * C1_ * sizeof(float));
  float*    d_p2  = (float*)    take((size_t)256 * 2 * C2_ * sizeof(float));
  float*    d_st1 = (float*)    take((size_t)2 * C1_ * sizeof(float));
  float*    d_st2 = (float*)    take((size_t)2 * C2_ * sizeof(float));

  // Weight conversion (tiny)
  cvt_f32_f16<<<(C1_ * CIN_ + 255) / 256, 256, 0, stream>>>(w1, d_W1h, C1_ * CIN_);
  cvt_f32_f16<<<(C2_ * C1_ + 255) / 256, 256, 0, stream>>>(w2, d_W2h, C2_ * C1_);

  // 3-NN + interpolation + concat (f16 activations)
  knn3_kernel<<<dim3(N_ / 256, B_), 256, 0, stream>>>(xyz1, xyz2, d_idx, d_wgt);
  interp_concat_kernel<<<NP_, 128, 0, stream>>>(points1, points2, d_idx, d_wgt, d_X1);

  // Layer 1: WMMA GEMM -> stats -> BN+ReLU -> f16
  gemm_f16_wmma<CIN_, C1_><<<dim3(NP_ / 64, C1_ / 64), 128, 0, stream>>>(d_X1, d_W1h, b1, d_H1);
  colstat_partial<C1_><<<256, C1_, 0, stream>>>(d_H1, d_p1);
  colstat_final<C1_><<<2 * C1_, 256, 0, stream>>>(d_p1, d_st1);
  bn_relu_to_f16<<<(NP_ * C1_ + 255) / 256, 256, 0, stream>>>(d_H1, d_st1, g1, be1, d_H1h,
                                                              C1_, NP_ * C1_);

  // Layer 2: WMMA GEMM directly into d_out -> stats -> BN+ReLU in place
  gemm_f16_wmma<C1_, C2_><<<dim3(NP_ / 64, C2_ / 64), 128, 0, stream>>>(d_H1h, d_W2h, b2, out);
  colstat_partial<C2_><<<256, C2_, 0, stream>>>(out, d_p2);
  colstat_final<C2_><<<2 * C2_, 256, 0, stream>>>(d_p2, d_st2);
  bn_relu_inplace_f32<<<(NP_ * C2_ + 255) / 256, 256, 0, stream>>>(out, d_st2, g2, be2,
                                                                   C2_, NP_ * C2_);
}
